// BuildingBlockEmbed_27255862460555
// MI455X (gfx1250) — matrix operand order repaired and checked by reading, not compiled, not run on hardware
//
#include <hip/hip_runtime.h>
#include <hip/hip_bf16.h>

typedef _Float16 h16;
typedef _Float16 v16h __attribute__((ext_vector_type(16)));
typedef _Float16 v8h  __attribute__((ext_vector_type(8)));
typedef float    v8f  __attribute__((ext_vector_type(8)));

#define NBLK    2048
#define NA      10
#define EPB     90
#define EPAD    96
#define NBASIS  16
#define TYPE_F  32
#define KQ      8          // key/query scalars
#define KV1     4          // key/query vectors
#define CO0     16         // output scalars
#define CO1     4          // output vectors
#define OUTF    28         // 16 + 4*3
#define NTHREADS 192       // 6 wave32 -> 6 edge tiles of 16

// ---- packed B-tile weight offsets (f16 elements); each N-tile = 32 lanes * 16 = 512
#define OFF_W1K0 0
#define OFF_W2K0 (OFF_W1K0 + 2 * 512)     // 24 tiles (nk0=384)
#define OFF_W1V0 (OFF_W2K0 + 24 * 512)
#define OFF_W2V0 (OFF_W1V0 + 2 * 512)     // 40 tiles (nv0=640)
#define OFF_W1K1 (OFF_W2V0 + 40 * 512)
#define OFF_W2K1 (OFF_W1K1 + 2 * 512)     // 15 tiles (nk1=240)
#define OFF_W1V1 (OFF_W2K1 + 15 * 512)
#define OFF_W2V1 (OFF_W1V1 + 2 * 512)     // 25 tiles (nv1=400)

// ------------------------------------------------------------------ helpers
__device__ __forceinline__ void ds_fence() {
  asm volatile("s_wait_dscnt 0" ::: "memory");
}

__device__ __forceinline__ float fast_rcp(float x) {
  return __builtin_amdgcn_rcpf(x);   // single v_rcp_f32, no div-fixup chain
}

__device__ __forceinline__ float silu_f(float x) {
  return x * fast_rcp(1.0f + __expf(-x));
}

// A operand, 16x32 f16: lane l -> row M = l&15, K = {klo..klo+7, 16+klo..16+klo+7},
// klo = 8*(l>>4).  `row` points at a 32-wide f16 row in LDS.
__device__ __forceinline__ v16h load_a_row(const h16* row, int klo) {
  v8h lo = *(const v8h*)(row + klo);
  v8h hi = *(const v8h*)(row + 16 + klo);
  return __builtin_shufflevector(lo, hi, 0, 1, 2, 3, 4, 5, 6, 7,
                                 8, 9, 10, 11, 12, 13, 14, 15);
}

// B operand, 32x16 f16, packed in ws as [ntile][lane][16] contiguous f16.
__device__ __forceinline__ v16h load_b_tile(const h16* wp, int nt, int lane) {
  return *(const v16h*)(wp + (size_t)(nt * 32 + lane) * 16);
}

__device__ __forceinline__ v8f wmma16(v16h a, v16h b) {
  v8f c = {};
  return __builtin_amdgcn_wmma_f32_16x16x32_f16(false, a, false, b,
                                                (short)0, c, false, false);
}

// ------------------------------------------------------------------ LDS layout
struct SM {
  float coords[NA][3];
  float s0[NA][TYPE_F];          // layer-0 input scalars
  float sA[NA][CO0];             // layer-0 output / layer-1 input
  float vA[NA][12];
  float sB[NA][CO0];             // layer-1 output
  float vB[NA][12];
  alignas(16) h16 embh[EPAD][32];    // radial basis (cols 16..31 zero)
  alignas(16) h16 hbuf[6 * 16][32];  // per-wave silu(hidden), f16
  float wbuf[6 * 16][16];            // per-wave 16x16 f32 GEMM tile
  float Y1[EPAD][3];
  float cut[EPAD];
  float dotl[EPAD];
  float al[EPAD];
  float vall[EPAD][OUTF];
  float qs[NA][KQ];
  float qv[NA][12];
  float mz[NA][2];
};

// ------------------------------------------------------------------ TP contraction
// Map the 16 GEMM output columns of this N-tile onto e3nn TP path segments.
template <int CI0, int CI1, int CO0_, int CO1_>
__device__ __forceinline__ void contract16(const float* wrow, int ntbase,
                                           const float* ssrc, const float* vdotv,
                                           const float* vsrc,
                                           float* os, float* ovc, float* ovd) {
  constexpr int SA = CI0 * CO0_;
  constexpr int SB = SA + CI1 * CO0_;
  constexpr int SC = SB + CI0 * CO1_;
#pragma unroll
  for (int c = 0; c < 16; ++c) {
    int col = ntbase + c;
    float wv = wrow[c];
    if (col < SA) {                                  // s * Y0 -> 0e
      os[col % CO0_] += ssrc[col / CO0_] * wv;
    } else if (CI1 > 0 && col < SB) {                // (v . Y1) -> 0e
      int cc = col - SA;
      os[cc % CO0_] += vdotv[cc / CO0_] * wv;
    } else if (col < SC) {                           // s * Y1 -> 1o (coeff)
      int cc = col - SB;
      ovc[cc % CO1_] += ssrc[cc / CO1_] * wv;
    } else if (CI1 > 0) {                            // v * Y0 -> 1o
      int cc = col - SC;
      int o = cc % CO1_, i = cc / CO1_;
      ovd[o * 3 + 0] += vsrc[i * 3 + 0] * wv;
      ovd[o * 3 + 1] += vsrc[i * 3 + 1] * wv;
      ovd[o * 3 + 2] += vsrc[i * 3 + 2] * wv;
    }
  }
}

// ------------------------------------------------------------------ one MLP+TP path
// 16-edge tile per wave:  w = silu(emb @ W1) @ W2, fused with TP contraction.
template <int CI0, int CI1, int CO0_, int CO1_, int NT2>
__device__ __forceinline__ void edge_path(SM* sm, int w, int lane,
                                          const h16* w1p, const h16* w2p,
                                          const float* ssrc, const float* vdotv,
                                          const float* vsrc,
                                          float* os, float* ovc, float* ovd) {
  int m0 = (lane >> 4) * 8;   // D-tile row base for this lane
  int n  = lane & 15;         // D-tile column for this lane
  // stage 1: hidden = silu(emb @ W1)   (K = 16 basis, zero-padded to 32)
  v16h a = load_a_row(sm->embh[w * 16 + (lane & 15)], (lane >> 4) * 8);
  v8f h0 = wmma16(a, load_b_tile(w1p, 0, lane));
  v8f h1 = wmma16(a, load_b_tile(w1p, 1, lane));
  h16* hb = &sm->hbuf[w * 16][0];
#pragma unroll
  for (int r = 0; r < 8; ++r) {
    hb[(m0 + r) * 32 + n]      = (h16)silu_f(h0[r]);
    hb[(m0 + r) * 32 + 16 + n] = (h16)silu_f(h1[r]);
  }
  ds_fence();
  // stage 2: w-tiles = hidden @ W2, consumed tile-by-tile
  v16h a2 = load_a_row(&hb[(lane & 15) * 32], (lane >> 4) * 8);
  float* wb = &sm->wbuf[w * 16][0];
  for (int nt = 0; nt < NT2; ++nt) {
    if (nt + 1 < NT2)  // keep the (L2-resident) weight stream ahead of the WMMAs
      __builtin_prefetch(w2p + (size_t)((nt + 1) * 32 + lane) * 16, 0, 0);
    v8f d = wmma16(a2, load_b_tile(w2p, nt, lane));
#pragma unroll
    for (int r = 0; r < 8; ++r) wb[(m0 + r) * 16 + n] = d[r];
    ds_fence();
    if (lane < 16) {
      contract16<CI0, CI1, CO0_, CO1_>(&wb[lane * 16], nt * 16,
                                       ssrc, vdotv, vsrc, os, ovc, ovd);
    }
  }
}

// ------------------------------------------------------------------ one attention layer
template <int CI0, int CI1>
__device__ void run_layer(SM* sm, int tid,
                          const float* sIn, int sStride, const float* vIn,
                          float (*sOut)[CO0], float (*vOut)[12],
                          const float* qsW, const float* qvW,
                          const h16* w1k, const h16* w2k,
                          const h16* w1v, const h16* w2v) {
  constexpr int NTK = (CI0 + CI1) * 12 / 16;  // key  N-tiles (24 / 15)
  constexpr int NTV = (CI0 + CI1) * 20 / 16;  // value N-tiles (40 / 25)

  // q projections
  for (int t = tid; t < NA * KQ; t += NTHREADS) {
    int atom = t / KQ, o = t % KQ;
    float acc = 0.f;
    for (int i = 0; i < CI0; ++i) acc += sIn[atom * sStride + i] * qsW[i * KQ + o];
    sm->qs[atom][o] = acc;
  }
  if (CI1 > 0) {
    for (int t = tid; t < NA * KV1 * 3; t += NTHREADS) {
      int atom = t / 12, rem = t % 12, o = rem / 3, x = rem % 3;
      float acc = 0.f;
      for (int c = 0; c < CI1; ++c) acc += vIn[atom * 12 + c * 3 + x] * qvW[c * KV1 + o];
      sm->qv[atom][o * 3 + x] = acc;
    }
  }
  __syncthreads();

  int w = tid >> 5, lane = tid & 31;
  int e = w * 16 + (lane & 15);
  bool ev = (lane < 16) && (e < EPB);
  int src = 0, dst = 0;
  if (ev) {
    int i = e / 9, jj = e % 9;
    src = i;
    dst = jj + (jj >= i ? 1 : 0);
  }
  float ssrc[CI0];
  float vsrc[CI1 > 0 ? CI1 * 3 : 1];
  float vdotv[CI1 > 0 ? CI1 : 1];
  float y[3] = {0.f, 0.f, 0.f};
  if (lane < 16) {
    y[0] = sm->Y1[e][0]; y[1] = sm->Y1[e][1]; y[2] = sm->Y1[e][2];
    for (int i = 0; i < CI0; ++i) ssrc[i] = ev ? sIn[src * sStride + i] : 0.f;
    if (CI1 > 0) {
      for (int c = 0; c < CI1; ++c) {
        float vx = ev ? vIn[src * 12 + c * 3 + 0] : 0.f;
        float vy = ev ? vIn[src * 12 + c * 3 + 1] : 0.f;
        float vz = ev ? vIn[src * 12 + c * 3 + 2] : 0.f;
        vsrc[c * 3 + 0] = vx; vsrc[c * 3 + 1] = vy; vsrc[c * 3 + 2] = vz;
        vdotv[c] = vx * y[0] + vy * y[1] + vz * y[2];
      }
    }
  } else {
    for (int i = 0; i < CI0; ++i) ssrc[i] = 0.f;
    if (CI1 > 0)
      for (int c = 0; c < CI1; ++c) {
        vsrc[c * 3] = vsrc[c * 3 + 1] = vsrc[c * 3 + 2] = 0.f;
        vdotv[c] = 0.f;
      }
  }

  // ---- key path + attention logits
  {
    float ks[KQ], kvc[KV1], kvd[KV1 * 3];
    for (int o = 0; o < KQ; ++o) ks[o] = 0.f;
    for (int o = 0; o < KV1; ++o) kvc[o] = 0.f;
    for (int o = 0; o < KV1 * 3; ++o) kvd[o] = 0.f;
    edge_path<CI0, CI1, KQ, KV1, NTK>(sm, w, lane, w1k, w2k,
                                      ssrc, vdotv, vsrc, ks, kvc, kvd);
    if (ev) {
      float dot = 0.f;
      for (int o = 0; o < KQ; ++o) dot += sm->qs[dst][o] * ks[o];
      if (CI1 > 0) {  // q_v is zero when the layer input has no vectors
        for (int o = 0; o < KV1; ++o)
          for (int x = 0; x < 3; ++x)
            dot += sm->qv[dst][o * 3 + x] * (kvc[o] * y[x] + kvd[o * 3 + x]);
      }
      sm->dotl[e] = dot;
    }
  }

  // ---- value path
  {
    float vs[CO0], vvc[CO1], vvd[CO1 * 3];
    for (int o = 0; o < CO0; ++o) vs[o] = 0.f;
    for (int o = 0; o < CO1; ++o) vvc[o] = 0.f;
    for (int o = 0; o < CO1 * 3; ++o) vvd[o] = 0.f;
    edge_path<CI0, CI1, CO0, CO1, NTV>(sm, w, lane, w1v, w2v,
                                       ssrc, vdotv, vsrc, vs, vvc, vvd);
    if (ev) {
      for (int f = 0; f < CO0; ++f) sm->vall[e][f] = vs[f];
      for (int o = 0; o < CO1; ++o)
        for (int x = 0; x < 3; ++x)
          sm->vall[e][CO0 + o * 3 + x] = vvc[o] * y[x] + vvd[o * 3 + x];
    }
  }
  __syncthreads();

  // ---- stable softmax over the 9 incoming edges of each atom
  if (tid < NA) {
    int j = tid;
    float m = -1e30f;
    for (int i = 0; i < NA; ++i) {
      if (i == j) continue;
      int ee = i * 9 + (j > i ? j - 1 : j);
      m = fmaxf(m, sm->dotl[ee]);
    }
    float z = 0.f;
    for (int i = 0; i < NA; ++i) {
      if (i == j) continue;
      int ee = i * 9 + (j > i ? j - 1 : j);
      z += sm->cut[ee] * __expf(sm->dotl[ee] - m);
    }
    sm->mz[j][0] = m;
    sm->mz[j][1] = z;
  }
  __syncthreads();
  for (int t = tid; t < EPB; t += NTHREADS) {
    int i = t / 9, jj = t % 9, j = jj + (jj >= i ? 1 : 0);
    float expv = sm->cut[t] * __expf(sm->dotl[t] - sm->mz[j][0]);
    float alpha = expv * fast_rcp(fmaxf(sm->mz[j][1], 1e-9f));
    sm->al[t] = sqrtf(fmaxf(alpha, 0.f));
  }
  __syncthreads();

  // ---- weighted value aggregation -> next features
  for (int t = tid; t < NA * OUTF; t += NTHREADS) {
    int atom = t / OUTF, f = t % OUTF;
    float acc = 0.f;
    for (int i = 0; i < NA; ++i) {
      if (i == atom) continue;
      int ee = i * 9 + (atom > i ? atom - 1 : atom);
      acc += sm->al[ee] * sm->vall[ee][f];
    }
    if (f < CO0) sOut[atom][f] = acc;
    else         vOut[atom][f - CO0] = acc;
  }
  __syncthreads();
}

// ------------------------------------------------------------------ weight packing
// Re-lays every W1/W2 into WMMA B-operand tiles (f16): [ntile][lane][16],
// lane l -> column n = ntile*16 + (l&15), K = 8*(l>=16) + e (e<8), 16 + 8*(l>=16) + e-8 (e>=8)  (zero-pad K>=Kreal).
__global__ void pack_weights(const float* kw10, const float* kw20,
                             const float* vw10, const float* vw20,
                             const float* kw11, const float* kw21,
                             const float* vw11, const float* vw21, h16* dst) {
  const float* srcs[8] = {kw10, kw20, vw10, vw20, kw11, kw21, vw11, vw21};
  const int Ks[8]   = {16, 32, 16, 32, 16, 32, 16, 32};
  const int NCs[8]  = {32, 384, 32, 640, 32, 240, 32, 400};
  const int offs[8] = {OFF_W1K0, OFF_W2K0, OFF_W1V0, OFF_W2V0,
                       OFF_W1K1, OFF_W2K1, OFF_W1V1, OFF_W2V1};
  int mi = blockIdx.x;
  const float* src = srcs[mi];
  int K = Ks[mi], NC = NCs[mi];
  h16* d = dst + offs[mi];
  int total = (NC / 16) * 512;
  for (int idx = threadIdx.x; idx < total; idx += blockDim.x) {
    int nt = idx >> 9;
    int rem = idx & 511;
    int lane = rem >> 4;
    int e = rem & 15;
    int k = ((e >> 3) << 3) + ((lane >> 4) << 3) + e;  // 0..31
    int col = nt * 16 + (lane & 15);
    float v = (k < K) ? src[k * NC + col] : 0.0f;
    d[(nt * 32 + lane) * 16 + e] = (h16)v;
  }
}

// ------------------------------------------------------------------ main kernel
__global__ void __launch_bounds__(NTHREADS, 1)
bbe_main(const int* __restrict__ atom_types, const float* __restrict__ coords_g,
         const float* __restrict__ embed_table,
         const float* __restrict__ qs0, const float* __restrict__ qs1,
         const float* __restrict__ qv1,
         const h16* __restrict__ wpack, float* __restrict__ out) {
  __shared__ SM sm;
  int blk = blockIdx.x, tid = threadIdx.x;

  // atom features + coordinates
  for (int t = tid; t < NA * 3; t += NTHREADS)
    sm.coords[t / 3][t % 3] = coords_g[(size_t)blk * NA * 3 + t];
  for (int t = tid; t < NA * TYPE_F; t += NTHREADS) {
    int atom = t / TYPE_F, i = t % TYPE_F;
    int ty = atom_types[blk * NA + atom] - 1;
    sm.s0[atom][i] = embed_table[ty * TYPE_F + i];
  }
  __syncthreads();

  // per-edge geometry, radial basis (f16), soft cutoff
  for (int e = tid; e < EPAD; e += NTHREADS) {
    if (e < EPB) {
      int i = e / 9, jj = e % 9, j = jj + (jj >= i ? 1 : 0);
      float rx = sm.coords[i][0] - sm.coords[j][0];
      float ry = sm.coords[i][1] - sm.coords[j][1];
      float rz = sm.coords[i][2] - sm.coords[j][2];
      float d = sqrtf(rx * rx + ry * ry + rz * rz);
      float inv = 1.732050808f * fast_rcp(fmaxf(d, 1e-9f));  // sqrt(3) * unit
      sm.Y1[e][0] = rx * inv; sm.Y1[e][1] = ry * inv; sm.Y1[e][2] = rz * inv;
      float x10 = 10.f * (1.f - d * 0.2f);
      sm.cut[e] = (x10 > 0.f) ? __expf(-fast_rcp(fmaxf(x10, 1e-9f))) : 0.f;
      const float step = 5.0f / 15.0f, invw = 16.0f / 5.0f;
      for (int b = 0; b < NBASIS; ++b) {
        float t2 = (d - step * (float)b) * invw;
        sm.embh[e][b] = (h16)(__expf(-t2 * t2) * 4.0f);  // * sqrt(NBASIS)
      }
      for (int b = NBASIS; b < 32; ++b) sm.embh[e][b] = (h16)0.f;
    } else {
      for (int b = 0; b < 32; ++b) sm.embh[e][b] = (h16)0.f;
      sm.Y1[e][0] = sm.Y1[e][1] = sm.Y1[e][2] = 0.f;
      sm.cut[e] = 0.f;
    }
  }
  __syncthreads();

  run_layer<TYPE_F, 0>(&sm, tid, &sm.s0[0][0], TYPE_F, nullptr, sm.sA, sm.vA,
                       qs0, nullptr,
                       wpack + OFF_W1K0, wpack + OFF_W2K0,
                       wpack + OFF_W1V0, wpack + OFF_W2V0);
  run_layer<CO0, CO1>(&sm, tid, &sm.sA[0][0], CO0, &sm.vA[0][0], sm.sB, sm.vB,
                      qs1, qv1,
                      wpack + OFF_W1K1, wpack + OFF_W2K1,
                      wpack + OFF_W1V1, wpack + OFF_W2V1);

  // global mean pool over atoms
  for (int f = tid; f < OUTF; f += NTHREADS) {
    float acc = 0.f;
    for (int atom = 0; atom < NA; ++atom)
      acc += (f < CO0) ? sm.sB[atom][f] : sm.vB[atom][f - CO0];
    out[(size_t)blk * OUTF + f] = acc * 0.1f;
  }
}

// ------------------------------------------------------------------ launch
extern "C" void kernel_launch(void* const* d_in, const int* in_sizes, int n_in,
                              void* d_out, int out_size, void* d_ws, size_t ws_size,
                              hipStream_t stream) {
  const int*   atom_types = (const int*)d_in[0];
  const float* coords     = (const float*)d_in[1];
  // d_in[2]: bb_num_vec (constant NA) - unused
  const float* embed      = (const float*)d_in[3];
  const float* qs0        = (const float*)d_in[4];
  // d_in[5]: q_v_0 has zero elements - unused
  const float* kw10 = (const float*)d_in[6];
  const float* kw20 = (const float*)d_in[7];
  const float* vw10 = (const float*)d_in[8];
  const float* vw20 = (const float*)d_in[9];
  const float* qs1  = (const float*)d_in[10];
  const float* qv1  = (const float*)d_in[11];
  const float* kw11 = (const float*)d_in[12];
  const float* kw21 = (const float*)d_in[13];
  const float* vw11 = (const float*)d_in[14];
  const float* vw21 = (const float*)d_in[15];

  h16* wpack = (h16*)d_ws;  // ~112 KB of packed f16 B-tiles

  pack_weights<<<8, 256, 0, stream>>>(kw10, kw20, vw10, vw20,
                                      kw11, kw21, vw11, vw21, wpack);
  bbe_main<<<NBLK, NTHREADS, 0, stream>>>(atom_types, coords, embed,
                                          qs0, qs1, qv1, wpack, (float*)d_out);
}
